// Net_13159779795462
// MI455X (gfx1250) — matrix operand order repaired and checked by reading, not compiled, not run on hardware
//
#include <hip/hip_runtime.h>

typedef _Float16 v16h __attribute__((ext_vector_type(16)));
typedef _Float16 v8h  __attribute__((ext_vector_type(8)));
typedef float    v8f  __attribute__((ext_vector_type(8)));

#define BATCH 32768
#define KDIM  784
#define NDIM  3000
#define NPAD  3072
#define NBLK  24      // 3072 / 128 column blocks
#define MBLK  256     // 32768 / 128 row blocks

// A-fragment (16x32 f16, ISA 7.12.2): lanes 0-15 hold K [0..7] and [16..23],
// lanes 16-31 hold K [8..15] and [24..31]  ->  kk = (lane&16) ? 8 : 0,
// chunks of 8 at kk and kk+16.
__device__ __forceinline__ v16h make_fragA(const _Float16* rowp, int kk) {
  v8h a = *(const v8h*)(rowp + kk);
  v8h b = *(const v8h*)(rowp + kk + 16);
  v16h r;
#pragma unroll
  for (int i = 0; i < 8; ++i) { r[i] = a[i]; r[i + 8] = b[i]; }
  return r;
}

// B-fragment (32x16 f16): same K order as A-fragment: lanes 0-15 hold
// K [0..7] and [16..23], lanes 16-31 hold K [8..15] and [24..31]
// -> chunks of 8 at kk and kk+16, kk = (lane&16) ? 8 : 0.
__device__ __forceinline__ v16h make_fragB(const _Float16* colp, int kb) {
  v8h a = *(const v8h*)(colp + kb);
  v8h b = *(const v8h*)(colp + kb + 16);
  v16h r;
#pragma unroll
  for (int i = 0; i < 8; ++i) { r[i] = a[i]; r[i + 8] = b[i]; }
  return r;
}

// PASS 1: accumulate per-feature sum / sum-of-squares of t = x @ sign(w1)^T
// PASS 2: t -> BN -> sign -> int8, fc2 partial dot with sign(w2), int atomics
template <int PASS>
__global__ __launch_bounds__(256) void gemm_kernel(
    const float* __restrict__ x, const float* __restrict__ w1,
    const float* __restrict__ scale1, const float* __restrict__ off1,
    const float* __restrict__ w2,
    float* __restrict__ pS, float* __restrict__ pQ, int* __restrict__ oint)
{
  const int nb = blockIdx.x, mb = blockIdx.y;
  const int tid  = threadIdx.x;
  const int lane = tid & 31, wave = tid >> 5;
  const int wm = wave & 3, wn = wave >> 2;      // 4 waves along M x 2 along N
  const int ln = lane & 15;
  const int kk = (lane & 16) ? 8 : 0;           // A-frag chunk select
  const int kb = (lane & 16) ? 16 : 0;          // unused: B-frag also uses kk

  __shared__ __align__(16) _Float16 Ah[128][40];  // x high f16
  __shared__ __align__(16) _Float16 Al[128][40];  // x low  f16 (residual)
  __shared__ __align__(16) _Float16 Bs[128][40];  // sign(w1) f16

  v8f acc[2][4] = {};

  const int row  = tid >> 1;   // 0..127: tile row (A) / tile col-feature (B)
  const int part = tid & 1;    // which 16-wide half of the 32-wide K tile

  for (int ks = 0; ks < 25; ++ks) {              // K padded 784 -> 800
    const int kbase = ks * 32 + part * 16;
    __syncthreads();
    // ---- stage A tile: f32 -> (hi, lo) f16 split -------------------------
    {
      float fv[16];
      if (kbase < KDIM) {
        const float4* p = (const float4*)(x + (size_t)(mb * 128 + row) * KDIM + kbase);
#pragma unroll
        for (int c = 0; c < 4; ++c) {
          float4 f = p[c];
          fv[4*c] = f.x; fv[4*c+1] = f.y; fv[4*c+2] = f.z; fv[4*c+3] = f.w;
        }
      } else {
#pragma unroll
        for (int c = 0; c < 16; ++c) fv[c] = 0.f;
      }
      v8h h0, h1, l0, l1;
#pragma unroll
      for (int c = 0; c < 8; ++c) {
        _Float16 a = (_Float16)fv[c];     h0[c] = a; l0[c] = (_Float16)(fv[c]     - (float)a);
        _Float16 b = (_Float16)fv[c + 8]; h1[c] = b; l1[c] = (_Float16)(fv[c + 8] - (float)b);
      }
      *(v8h*)&Ah[row][part * 16]     = h0;
      *(v8h*)&Ah[row][part * 16 + 8] = h1;
      *(v8h*)&Al[row][part * 16]     = l0;
      *(v8h*)&Al[row][part * 16 + 8] = l1;
    }
    // ---- stage B tile: sign(w1) -> +-1 f16 -------------------------------
    {
      v8h s0 = {}, s1 = {};
      if (kbase < KDIM) {
        const int wr = nb * 128 + row;
        const int cr = wr < NDIM ? wr : NDIM - 1;   // clamp for memory safety
        const float4* p = (const float4*)(w1 + (size_t)cr * KDIM + kbase);
        float fv[16];
#pragma unroll
        for (int c = 0; c < 4; ++c) {
          float4 f = p[c];
          fv[4*c] = f.x; fv[4*c+1] = f.y; fv[4*c+2] = f.z; fv[4*c+3] = f.w;
        }
#pragma unroll
        for (int c = 0; c < 8; ++c) {
          s0[c] = (_Float16)((fv[c]   > 0.f) ? 1.f : ((fv[c]   < 0.f) ? -1.f : 0.f));
          s1[c] = (_Float16)((fv[c+8] > 0.f) ? 1.f : ((fv[c+8] < 0.f) ? -1.f : 0.f));
        }
      }
      *(v8h*)&Bs[row][part * 16]     = s0;
      *(v8h*)&Bs[row][part * 16 + 8] = s1;
    }
    __syncthreads();

    // ---- WMMA: 2 M-tiles x 4 N-tiles, hi + lo pass -> 16 wmma / wave -----
    v16h afh[2], afl[2], bf[4];
#pragma unroll
    for (int tm = 0; tm < 2; ++tm) {
      const int ar = wm * 32 + tm * 16 + ln;
      afh[tm] = make_fragA(&Ah[ar][0], kk);
      afl[tm] = make_fragA(&Al[ar][0], kk);
    }
#pragma unroll
    for (int tn = 0; tn < 4; ++tn) {
      const int bc = wn * 64 + tn * 16 + ln;
      bf[tn] = make_fragB(&Bs[bc][0], kk);
    }
#pragma unroll
    for (int tm = 0; tm < 2; ++tm)
#pragma unroll
      for (int tn = 0; tn < 4; ++tn) {
        acc[tm][tn] = __builtin_amdgcn_wmma_f32_16x16x32_f16(
            false, afh[tm], false, bf[tn], (short)0, acc[tm][tn], false, false);
        acc[tm][tn] = __builtin_amdgcn_wmma_f32_16x16x32_f16(
            false, afl[tm], false, bf[tn], (short)0, acc[tm][tn], false, false);
      }
  }

  if constexpr (PASS == 1) {
    // Deterministic fixed-order column reduction: 16 contributor slots/col.
    __shared__ float redS[16][128];
    __shared__ float redQ[16][128];
    const int half = lane >> 4;
#pragma unroll
    for (int tn = 0; tn < 4; ++tn) {
      const int col = wn * 64 + tn * 16 + ln;
#pragma unroll
      for (int tm = 0; tm < 2; ++tm) {
        float s = 0.f, q = 0.f;
#pragma unroll
        for (int v = 0; v < 8; ++v) { float t = acc[tm][tn][v]; s += t; q += t * t; }
        const int ctr = wm * 4 + tm * 2 + half;
        redS[ctr][col] = s;
        redQ[ctr][col] = q;
      }
    }
    __syncthreads();
    if (tid < 128) {
      float S = 0.f, Q = 0.f;
#pragma unroll
      for (int c = 0; c < 16; ++c) { S += redS[c][tid]; Q += redQ[c][tid]; }
      const int gcol = nb * 128 + tid;
      pS[(size_t)mb * NPAD + gcol] = S;
      pQ[(size_t)mb * NPAD + gcol] = Q;
    }
  } else {
    __shared__ signed char hbs[128][132];    // binarized activations (tile)
    __shared__ signed char s2s[10][128];     // sign(w2) slice
    for (int i = tid; i < 1280; i += 256) {
      const int k = i >> 7, j = i & 127;
      const int gj = nb * 128 + j;
      const float wv = (gj < NDIM) ? w2[k * NDIM + gj] : 0.f;   // 0 masks pad cols
      s2s[k][j] = (wv > 0.f) ? 1 : ((wv < 0.f) ? -1 : 0);
    }
    const int half = lane >> 4;
#pragma unroll
    for (int tn = 0; tn < 4; ++tn) {
      const int lcol = wn * 64 + tn * 16 + ln;
      const int gj = nb * 128 + lcol;
      const float sc = scale1[gj];
      const float of = off1[gj];
#pragma unroll
      for (int tm = 0; tm < 2; ++tm) {
        const int rb = wm * 32 + tm * 16 + half * 8;   // C layout: M = v + 8*half
#pragma unroll
        for (int v = 0; v < 8; ++v) {
          const float bnv = sc * acc[tm][tn][v] + of;  // sign(clip(BN)) == sign(BN)
          hbs[rb + v][lcol] = (bnv > 0.f) ? 1 : ((bnv < 0.f) ? -1 : 0);
        }
      }
    }
    __syncthreads();
    // fc2 partial: exact integer dot, order-independent int atomics.
    {
      const int r  = tid >> 1;
      const int kh = (tid & 1) * 5;
      int a0 = 0, a1 = 0, a2 = 0, a3 = 0, a4 = 0;
#pragma unroll 4
      for (int j = 0; j < 128; ++j) {
        const int h = hbs[r][j];
        a0 += h * s2s[kh + 0][j];
        a1 += h * s2s[kh + 1][j];
        a2 += h * s2s[kh + 2][j];
        a3 += h * s2s[kh + 3][j];
        a4 += h * s2s[kh + 4][j];
      }
      int* orow = oint + (size_t)(mb * 128 + r) * 10 + kh;
      atomicAdd(orow + 0, a0);
      atomicAdd(orow + 1, a1);
      atomicAdd(orow + 2, a2);
      atomicAdd(orow + 3, a3);
      atomicAdd(orow + 4, a4);
    }
  }
}

__global__ __launch_bounds__(256) void finalize1_kernel(
    const float* __restrict__ pS, const float* __restrict__ pQ,
    const float* __restrict__ g1, const float* __restrict__ be1,
    float* __restrict__ scale1, float* __restrict__ off1)
{
  const int j = blockIdx.x * 256 + threadIdx.x;   // 0..3071
  double S = 0.0, Q = 0.0;
  for (int mb = 0; mb < MBLK; ++mb) {
    S += (double)pS[(size_t)mb * NPAD + j];
    Q += (double)pQ[(size_t)mb * NPAD + j];
  }
  const double m   = S / (double)BATCH;
  const double var = Q / (double)BATCH - m * m;
  float sc = 0.f, of = 0.f;
  if (j < NDIM) {
    sc = (float)((double)g1[j] / sqrt(var + 1e-5));   // b1 cancels inside BN
    of = (float)((double)be1[j] - (double)sc * m);
  }
  scale1[j] = sc;
  off1[j]   = of;
}

__global__ __launch_bounds__(256) void finalize2_kernel(
    const int* __restrict__ oint, const float* __restrict__ g2,
    const float* __restrict__ be2, float* __restrict__ scale2,
    float* __restrict__ off2)
{
  __shared__ long long rs[256][10];
  __shared__ long long rq[256][10];
  long long s[10], q[10];
#pragma unroll
  for (int k = 0; k < 10; ++k) { s[k] = 0; q[k] = 0; }
  for (int i = threadIdx.x; i < BATCH; i += 256) {
    const int* r = oint + (size_t)i * 10;
#pragma unroll
    for (int k = 0; k < 10; ++k) { const long long v = r[k]; s[k] += v; q[k] += v * v; }
  }
#pragma unroll
  for (int k = 0; k < 10; ++k) { rs[threadIdx.x][k] = s[k]; rq[threadIdx.x][k] = q[k]; }
  __syncthreads();
  if (threadIdx.x < 10) {
    const int k = threadIdx.x;
    long long S = 0, Q = 0;
    for (int t = 0; t < 256; ++t) { S += rs[t][k]; Q += rq[t][k]; }
    const double m   = (double)S / (double)BATCH;
    const double var = (double)Q / (double)BATCH - m * m;   // b2 cancels
    const float sc = (float)((double)g2[k] / sqrt(var + 1e-5));
    scale2[k] = sc;
    off2[k]   = (float)((double)be2[k] - (double)sc * m);
  }
}

__global__ __launch_bounds__(256) void apply_kernel(
    const int* __restrict__ oint, const float* __restrict__ scale2,
    const float* __restrict__ off2, float* __restrict__ out)
{
  const int idx = blockIdx.x * 256 + threadIdx.x;
  if (idx < BATCH * 10) {
    const int k = idx % 10;
    out[idx] = scale2[k] * (float)oint[idx] + off2[k];
  }
}

extern "C" void kernel_launch(void* const* d_in, const int* in_sizes, int n_in,
                              void* d_out, int out_size, void* d_ws, size_t ws_size,
                              hipStream_t stream)
{
  (void)in_sizes; (void)n_in; (void)out_size; (void)ws_size;
  const float* x   = (const float*)d_in[0];
  const float* w1  = (const float*)d_in[1];
  // d_in[2] = b1 (cancels in BN), d_in[6] = b2 (cancels in BN)
  const float* g1  = (const float*)d_in[3];
  const float* be1 = (const float*)d_in[4];
  const float* w2  = (const float*)d_in[5];
  const float* g2  = (const float*)d_in[7];
  const float* be2 = (const float*)d_in[8];
  float* out = (float*)d_out;

  // workspace layout (~7.6 MB total)
  float* pS     = (float*)d_ws;                       // 256*3072
  float* pQ     = pS + (size_t)MBLK * NPAD;           // 256*3072
  float* scale1 = pQ + (size_t)MBLK * NPAD;           // 3072
  float* off1   = scale1 + NPAD;                      // 3072
  int*   oint   = (int*)(off1 + NPAD);                // 32768*10
  float* scale2 = (float*)(oint + BATCH * 10);        // 10
  float* off2   = scale2 + 10;                        // 10

  hipMemsetAsync(oint, 0, sizeof(int) * BATCH * 10, stream);

  gemm_kernel<1><<<dim3(NBLK, MBLK), 256, 0, stream>>>(
      x, w1, nullptr, nullptr, nullptr, pS, pQ, nullptr);
  finalize1_kernel<<<NPAD / 256, 256, 0, stream>>>(pS, pQ, g1, be1, scale1, off1);
  gemm_kernel<2><<<dim3(NBLK, MBLK), 256, 0, stream>>>(
      x, w1, scale1, off1, w2, nullptr, nullptr, oint);
  finalize2_kernel<<<1, 256, 0, stream>>>(oint, g2, be2, scale2, off2);
  apply_kernel<<<(BATCH * 10 + 255) / 256, 256, 0, stream>>>(oint, scale2, off2, out);
}
